// HeteroRelGAT_15556371546276
// MI455X (gfx1250) — matrix operand order repaired and checked by reading, not compiled, not run on hardware
//
#include <hip/hip_runtime.h>
#include <hip/hip_bf16.h>
#include <stdint.h>

// ---------------------------------------------------------------------------
// HeteroRelGAT for MI455X (gfx1250): bf16 WMMA GEMMs + L2-resident atomic
// edge softmax/scatter. wave32; 16x16x32 bf16 WMMA with f32 accumulation.
// ---------------------------------------------------------------------------

namespace {

constexpr int kNSrc  = 100000;
constexpr int kNDst  = 50000;
constexpr int kE     = 600000;
constexpr int kInDim = 128;
constexpr int kHid   = 32;
constexpr int kHeads = 4;
constexpr int kFeat  = kHid * kHeads;   // 128
constexpr int kOut   = 64;

typedef __attribute__((ext_vector_type(8)))  float  v8f;
typedef __attribute__((ext_vector_type(16))) __bf16 v16bf;

// Native fptrunc (RTNE) -> lets the backend pick v_cvt_*bf16* on gfx1250
// instead of a 3-op integer emulation sequence.
__device__ __forceinline__ __bf16 f2bf(float f) { return (__bf16)f; }

// order-preserving float<->uint mapping for atomicMax on floats
__device__ __forceinline__ unsigned f2ord(float f) {
  unsigned u = __float_as_uint(f);
  return (u & 0x80000000u) ? ~u : (u | 0x80000000u);
}
__device__ __forceinline__ float ord2f(unsigned u) {
  return __uint_as_float((u & 0x80000000u) ? (u & 0x7FFFFFFFu) : ~u);
}

// A-fragment (16x32 bf16, M x K) per ISA layout:
//   lanes 0-15 : row = lane,    elems 0-7 = K 0-7,  elems 8-15 = K 16-23
//   lanes 16-31: row = lane-16, elems 0-7 = K 8-15, elems 8-15 = K 24-31
__device__ __forceinline__ v16bf load_a_bf(const __bf16* __restrict__ rowp, int kb) {
  v16bf a;
#pragma unroll
  for (int i = 0; i < 8; ++i) { a[i] = rowp[kb + i]; a[8 + i] = rowp[kb + 16 + i]; }
  return a;
}
__device__ __forceinline__ v16bf load_a_f32(const float* __restrict__ rowp, int kb) {
  v16bf a;
#pragma unroll
  for (int i = 0; i < 8; ++i) { a[i] = f2bf(rowp[kb + i]); a[8 + i] = f2bf(rowp[kb + 16 + i]); }
  return a;
}

// ---------------------------------------------------------------------------
// Pack f32 B (K x N row-major) into WMMA B-fragment-major bf16:
// fragment (kt,nt): lane L<16 holds col nt*16+L, K rows kt*32+0..7,16..23;
// lanes 16-31 hold col nt*16+L-16, K rows kt*32+8..15,24..31. 16 bf16 per lane,
// stored contiguously -> one 32B load per lane in the GEMM kernels.
// ---------------------------------------------------------------------------
__global__ void pack_b_kernel(const float* __restrict__ B, int ldb, int nt_count,
                              __bf16* __restrict__ Bp) {
  int frag = blockIdx.x;
  int lane = threadIdx.x;                       // 32 threads
  int kt = frag / nt_count, nt = frag % nt_count;
  int col = nt * 16 + (lane & 15);
  int ks  = kt * 32 + ((lane >> 4) ? 8 : 0);
  __bf16* dst = Bp + ((size_t)frag * 32 + lane) * 16;
#pragma unroll
  for (int i = 0; i < 16; ++i) dst[i] = f2bf(B[(size_t)(ks + i + (i >= 8 ? 8 : 0)) * ldb + col]);
}

// ---------------------------------------------------------------------------
// h = x @ Wp + bp   (M x 128) @ (128 x 32); writes f32 and bf16 copies.
// One wave per 16-row tile; KT=4, NT=2 -> 8 WMMAs per wave.
// ---------------------------------------------------------------------------
__global__ void proj_kernel(const float* __restrict__ X, const __bf16* __restrict__ Bp,
                            const float* __restrict__ bias,
                            float* __restrict__ H, __bf16* __restrict__ Hbf, int M) {
  int lane = threadIdx.x & 31;
  int wave = threadIdx.x >> 5;
  int row0 = (blockIdx.x * (blockDim.x >> 5) + wave) * 16;
  if (row0 >= M) return;
  const v16bf* bp = (const v16bf*)Bp;
  int r  = row0 + (lane & 15);
  int kb = (lane >> 4) ? 8 : 0;
  const float* xr = X + (size_t)r * kInDim;
  v8f c[2] = {};
#pragma unroll
  for (int kt = 0; kt < 4; ++kt) {
    v16bf a = load_a_f32(xr + kt * 32, kb);
#pragma unroll
    for (int nt = 0; nt < 2; ++nt) {
      v16bf b = bp[(kt * 2 + nt) * 32 + lane];
      c[nt] = __builtin_amdgcn_wmma_f32_16x16x32_bf16(false, a, false, b,
                                                      (short)0, c[nt], false, false);
    }
  }
  int col = lane & 15;
  int rb  = row0 + ((lane >> 4) ? 8 : 0);
#pragma unroll
  for (int nt = 0; nt < 2; ++nt)
#pragma unroll
    for (int j = 0; j < 8; ++j) {
      float v = c[nt][j] + bias[nt * 16 + col];
      size_t off = (size_t)(rb + j) * kHid + nt * 16 + col;
      H[off] = v;
      Hbf[off] = f2bf(v);
    }
}

// ---------------------------------------------------------------------------
// fsrc = h_bf16 @ W_src   (M x 32) @ (32 x 128); KT=1, NT=8.
// ---------------------------------------------------------------------------
__global__ void feat_kernel(const __bf16* __restrict__ Hbf, const __bf16* __restrict__ Bp,
                            float* __restrict__ F, int M) {
  int lane = threadIdx.x & 31;
  int wave = threadIdx.x >> 5;
  int row0 = (blockIdx.x * (blockDim.x >> 5) + wave) * 16;
  if (row0 >= M) return;
  const v16bf* bp = (const v16bf*)Bp;
  int r  = row0 + (lane & 15);
  int kb = (lane >> 4) ? 8 : 0;
  v16bf a = load_a_bf(Hbf + (size_t)r * kHid, kb);
  v8f c[8] = {};
#pragma unroll
  for (int nt = 0; nt < 8; ++nt) {
    v16bf b = bp[nt * 32 + lane];
    c[nt] = __builtin_amdgcn_wmma_f32_16x16x32_bf16(false, a, false, b,
                                                    (short)0, c[nt], false, false);
  }
  int col = lane & 15;
  int rb  = row0 + ((lane >> 4) ? 8 : 0);
#pragma unroll
  for (int nt = 0; nt < 8; ++nt)
#pragma unroll
    for (int j = 0; j < 8; ++j)
      F[(size_t)(rb + j) * kFeat + nt * 16 + col] = c[nt][j];
}

// ---------------------------------------------------------------------------
// out = loc_bf16 @ Wf[:128,:] + bf  (comb's second half is zeros)
// (M x 128) @ (128 x 64); KT=4, NT=4.
// ---------------------------------------------------------------------------
__global__ void final_kernel(const __bf16* __restrict__ Loc, const __bf16* __restrict__ Bp,
                             const float* __restrict__ bias, float* __restrict__ Out, int M) {
  int lane = threadIdx.x & 31;
  int wave = threadIdx.x >> 5;
  int row0 = (blockIdx.x * (blockDim.x >> 5) + wave) * 16;
  if (row0 >= M) return;
  const v16bf* bp = (const v16bf*)Bp;
  int r  = row0 + (lane & 15);
  int kb = (lane >> 4) ? 8 : 0;
  const __bf16* lr = Loc + (size_t)r * kFeat;
  v8f c[4] = {};
#pragma unroll
  for (int kt = 0; kt < 4; ++kt) {
    v16bf a = load_a_bf(lr + kt * 32, kb);
#pragma unroll
    for (int nt = 0; nt < 4; ++nt) {
      v16bf b = bp[(kt * 4 + nt) * 32 + lane];
      c[nt] = __builtin_amdgcn_wmma_f32_16x16x32_bf16(false, a, false, b,
                                                      (short)0, c[nt], false, false);
    }
  }
  int col = lane & 15;
  int rb  = row0 + ((lane >> 4) ? 8 : 0);
#pragma unroll
  for (int nt = 0; nt < 4; ++nt)
#pragma unroll
    for (int j = 0; j < 8; ++j)
      Out[(size_t)(rb + j) * kOut + nt * 16 + col] = c[nt][j] + bias[nt * 16 + col];
}

// fold[k,h] = sum_d W[k, h*32+d] * a[h,d]   (collapses el/er to 32x4 GEMV)
__global__ void fold_kernel(const float* __restrict__ W, const float* __restrict__ a,
                            float* __restrict__ out) {
  int t = threadIdx.x;                      // 128 threads
  int k = t >> 2, h = t & 3;
  float s = 0.f;
#pragma unroll
  for (int d = 0; d < kHid; ++d) s += W[(size_t)k * kFeat + h * kHid + d] * a[h * kHid + d];
  out[k * kHeads + h] = s;
}

// score[n,h] = h[n,:] . fold[:,h]
__global__ void score_kernel(const float* __restrict__ H, const float* __restrict__ fold,
                             float* __restrict__ out, int n4) {
  int i = blockIdx.x * blockDim.x + threadIdx.x;
  if (i >= n4) return;
  int node = i >> 2, h = i & 3;
  const float* hr = H + (size_t)node * kHid;
  float s = 0.f;
#pragma unroll
  for (int k = 0; k < kHid; ++k) s += hr[k] * fold[k * kHeads + h];
  out[i] = s;
}

__global__ void fill_u32_kernel(unsigned* __restrict__ p, unsigned v, int n) {
  int i = blockIdx.x * blockDim.x + threadIdx.x;
  if (i < n) p[i] = v;
}

// e = leaky_relu(el[src]+er[dst]); segment-max via ordered-uint atomicMax
__global__ void edge_max_kernel(const int* __restrict__ src, const int* __restrict__ dst,
                                const float* __restrict__ el, const float* __restrict__ er,
                                float* __restrict__ ebuf, unsigned* __restrict__ emax, int n) {
  int i = blockIdx.x * blockDim.x + threadIdx.x;
  if (i >= n) return;
  int e = i >> 2, h = i & 3;
  float v = el[src[e] * 4 + h] + er[dst[e] * 4 + h];
  v = (v >= 0.f) ? v : 0.2f * v;
  ebuf[i] = v;
  atomicMax(&emax[dst[e] * 4 + h], f2ord(v));
}

// p = exp(e - emax[dst]); denom[dst] += p
__global__ void edge_exp_kernel(const int* __restrict__ dst, const unsigned* __restrict__ emax,
                                float* __restrict__ ebuf, float* __restrict__ denom, int n) {
  int i = blockIdx.x * blockDim.x + threadIdx.x;
  if (i >= n) return;
  int e = i >> 2, h = i & 3;
  float m = ord2f(emax[dst[e] * 4 + h]);
  float p = __expf(ebuf[i] - m);
  ebuf[i] = p;
  atomicAdd(&denom[dst[e] * 4 + h], p);
}

// acc[dst, c] += fsrc[src, c] * alpha  (one thread per edge-channel;
// acc is 25 MB -> lives in the 192 MB L2, atomics resolve in L2)
__global__ void edge_msg_kernel(const int* __restrict__ src, const int* __restrict__ dst,
                                const float* __restrict__ ebuf, const float* __restrict__ denom,
                                const float* __restrict__ F, float* __restrict__ acc, int n) {
  int i = blockIdx.x * blockDim.x + threadIdx.x;
  if (i >= n) return;
  int e = i >> 7, c = i & 127, h = c >> 5;
  int s = src[e], d = dst[e];
  float alpha = ebuf[e * 4 + h] / (denom[d * 4 + h] + 1e-9f);
  atomicAdd(&acc[(size_t)d * kFeat + c], F[(size_t)s * kFeat + c] * alpha);
}

// loc = relu(layernorm(0.5*acc + tile(hd,4)) * g + b), stored bf16
__global__ void combine_kernel(const float* __restrict__ acc, const float* __restrict__ H,
                               const float* __restrict__ g, const float* __restrict__ bta,
                               __bf16* __restrict__ loc) {
  int node = blockIdx.x, c = threadIdx.x;   // 128 threads
  float x = 0.5f * acc[(size_t)node * kFeat + c] + H[(size_t)node * kHid + (c & 31)];
  __shared__ float red[128];
  red[c] = x; __syncthreads();
#pragma unroll
  for (int s = 64; s > 0; s >>= 1) { if (c < s) red[c] += red[c + s]; __syncthreads(); }
  float mu = red[0] * (1.f / 128.f);
  __syncthreads();
  float dx = x - mu;
  red[c] = dx * dx; __syncthreads();
#pragma unroll
  for (int s = 64; s > 0; s >>= 1) { if (c < s) red[c] += red[c + s]; __syncthreads(); }
  float var = red[0] * (1.f / 128.f);
  float y = dx * rsqrtf(var + 1e-5f) * g[c] + bta[c];
  loc[(size_t)node * kFeat + c] = f2bf(fmaxf(y, 0.f));
}

}  // namespace

// ---------------------------------------------------------------------------
extern "C" void kernel_launch(void* const* d_in, const int* in_sizes, int n_in,
                              void* d_out, int out_size, void* d_ws, size_t ws_size,
                              hipStream_t stream) {
  (void)in_sizes; (void)n_in; (void)out_size; (void)ws_size;

  const float* x[2]    = {(const float*)d_in[0], (const float*)d_in[1]};
  const float* Wp[2]   = {(const float*)d_in[2], (const float*)d_in[8]};
  const float* bp[2]   = {(const float*)d_in[3], (const float*)d_in[9]};
  const float* ln_g[2] = {(const float*)d_in[4], (const float*)d_in[10]};
  const float* ln_b[2] = {(const float*)d_in[5], (const float*)d_in[11]};
  const float* Wf[2]   = {(const float*)d_in[6], (const float*)d_in[12]};
  const float* bfv[2]  = {(const float*)d_in[7], (const float*)d_in[13]};

  // relations ui,iu,uu,ii at indices 14 + r*6: W_src, W_dst, a_l, a_r, src, dst
  const float* W_src[4]; const float* W_dst[4];
  const float* a_l[4];   const float* a_r[4];
  const int*   e_src[4]; const int*   e_dst[4];
  for (int r = 0; r < 4; ++r) {
    int b = 14 + r * 6;
    W_src[r] = (const float*)d_in[b + 0];
    W_dst[r] = (const float*)d_in[b + 1];
    a_l[r]   = (const float*)d_in[b + 2];
    a_r[r]   = (const float*)d_in[b + 3];
    e_src[r] = (const int*)d_in[b + 4];
    e_dst[r] = (const int*)d_in[b + 5];
  }
  // relation r: src node-type, dst node-type (0=user,1=item); acc target = dst type
  const int relSrc[4] = {0, 1, 0, 1};
  const int relDst[4] = {1, 0, 0, 1};

  // ---- workspace carve (256B aligned) ----
  char* cur = (char*)d_ws;
  auto carve = [&](size_t bytes) -> void* {
    void* p = (void*)cur;
    cur += (bytes + 255) & ~(size_t)255;
    return p;
  };
  float*  h_f[2];  h_f[0]  = (float*)carve((size_t)kNSrc * kHid * 4);
                   h_f[1]  = (float*)carve((size_t)kNSrc * kHid * 4);
  __bf16* h_bf[2]; h_bf[0] = (__bf16*)carve((size_t)kNSrc * kHid * 2);
                   h_bf[1] = (__bf16*)carve((size_t)kNSrc * kHid * 2);
  float*  acc[2];  acc[0]  = (float*)carve((size_t)kNDst * kFeat * 4);
                   acc[1]  = (float*)carve((size_t)kNDst * kFeat * 4);
  float*  fsrc    = (float*)carve((size_t)kNSrc * kFeat * 4);
  float*  el      = (float*)carve((size_t)kNSrc * kHeads * 4);
  float*  er      = (float*)carve((size_t)kNDst * kHeads * 4);
  unsigned* emax  = (unsigned*)carve((size_t)kNDst * kHeads * 4);
  float*  denom   = (float*)carve((size_t)kNDst * kHeads * 4);
  float*  ebuf    = (float*)carve((size_t)kE * kHeads * 4);
  __bf16* loc     = (__bf16*)carve((size_t)kNDst * kFeat * 2);
  __bf16* wpP[2]; wpP[0] = (__bf16*)carve(8 * 512 * 2);
                  wpP[1] = (__bf16*)carve(8 * 512 * 2);
  __bf16* wsP     = (__bf16*)carve(8 * 512 * 2);
  __bf16* wfP     = (__bf16*)carve(16 * 512 * 2);
  float*  foldL   = (float*)carve(kHid * kHeads * 4);
  float*  foldR   = (float*)carve(kHid * kHeads * 4);

  const int projBlocks  = (kNSrc / 16 + 3) / 4;        // 4 waves/block
  const int finalBlocks = (kNDst / 16 + 3) / 4;
  const int e4  = kE * kHeads;                         // 2.4M
  const int e4B = (e4 + 255) / 256;
  const int eC  = kE * kFeat;                          // 76.8M
  const int eCB = (eC + 255) / 256;
  const int dh  = kNDst * kHeads;                      // 200k
  const int dhB = (dh + 255) / 256;

  // ---- projections: h = x @ Wp + bp (f32 + bf16 copies) ----
  for (int t = 0; t < 2; ++t) {
    pack_b_kernel<<<8, 32, 0, stream>>>(Wp[t], kHid, 2, wpP[t]);          // KT=4,NT=2
    proj_kernel<<<projBlocks, 128, 0, stream>>>(x[t], wpP[t], bp[t],
                                                h_f[t], h_bf[t], kNSrc);
  }
  // zero message accumulators
  fill_u32_kernel<<<(kNDst * kFeat + 255) / 256, 256, 0, stream>>>((unsigned*)acc[0], 0u, kNDst * kFeat);
  fill_u32_kernel<<<(kNDst * kFeat + 255) / 256, 256, 0, stream>>>((unsigned*)acc[1], 0u, kNDst * kFeat);

  // ---- relations (sequential, buffers reused) ----
  for (int r = 0; r < 4; ++r) {
    const float*  hs_f  = h_f[relSrc[r]];
    const __bf16* hs_bf = h_bf[relSrc[r]];
    const float*  hd_f  = h_f[relDst[r]];
    float*        ac    = acc[relDst[r]];

    pack_b_kernel<<<8, 32, 0, stream>>>(W_src[r], kFeat, 8, wsP);         // KT=1,NT=8
    fold_kernel<<<1, 128, 0, stream>>>(W_src[r], a_l[r], foldL);
    fold_kernel<<<1, 128, 0, stream>>>(W_dst[r], a_r[r], foldR);
    score_kernel<<<(kNSrc * kHeads + 255) / 256, 256, 0, stream>>>(hs_f, foldL, el, kNSrc * kHeads);
    score_kernel<<<dhB, 256, 0, stream>>>(hd_f, foldR, er, dh);
    feat_kernel<<<projBlocks, 128, 0, stream>>>(hs_bf, wsP, fsrc, kNSrc);

    fill_u32_kernel<<<dhB, 256, 0, stream>>>(emax, 0x00800000u, dh);      // ord(-FLT_MAX)
    fill_u32_kernel<<<dhB, 256, 0, stream>>>((unsigned*)denom, 0u, dh);

    edge_max_kernel<<<e4B, 256, 0, stream>>>(e_src[r], e_dst[r], el, er, ebuf, emax, e4);
    edge_exp_kernel<<<e4B, 256, 0, stream>>>(e_dst[r], emax, ebuf, denom, e4);
    edge_msg_kernel<<<eCB, 256, 0, stream>>>(e_src[r], e_dst[r], ebuf, denom, fsrc, ac, eC);
  }

  // ---- combine (mean + residual + layernorm + relu) and final GEMM ----
  float* out_f = (float*)d_out;
  for (int t = 0; t < 2; ++t) {
    combine_kernel<<<kNDst, 128, 0, stream>>>(acc[t], h_f[t], ln_g[t], ln_b[t], loc);
    pack_b_kernel<<<16, 32, 0, stream>>>(Wf[t], kOut, 4, wfP);            // KT=4,NT=4 (rows 0..127)
    final_kernel<<<finalBlocks, 128, 0, stream>>>(loc, wfP, bfv[t],
                                                  out_f + (size_t)t * kNDst * kOut, kNDst);
  }
}
